// AttentionCritic_31610959298717
// MI455X (gfx1250) — compile-verified, matrix operand order checked by reading, not run on hardware
//
#include <hip/hip_runtime.h>

// ---------------------------------------------------------------------------
// Fused AttentionCritic forward for MI455X (gfx1250, wave32, WMMA).
//
// Live dataflow (rest of the reference is dead code — hard_weights is unused
// and softmax over a singleton axis == 1.0):
//   x     = relu(obs @ enc_w^T + enc_b)            [65535,256] -> [65535,128]
//   h_out = GRUCell(x, hidden; gru_w_ih, gru_w_hh) [65535,128]
//   v     = relu(h_out @ v_w^T + v_b)              [65535,64]
//   out   = [h_out, v] @ dec_w^T + dec_b           [65535,10]
//   return (out, h_out)
//
// All GEMMs via v_wmma_f32_16x16x32_f16 (f16 A/B, f32 accum). Weights are
// pre-converted to f16 in d_ws (dec_w zero-padded to 16 rows). Each wave
// owns 16 batch rows; 8 waves / 256 threads per block -> 128 rows per block.
// ---------------------------------------------------------------------------

typedef __attribute__((ext_vector_type(16))) _Float16 v16h;
typedef __attribute__((ext_vector_type(8)))  _Float16 v8h;
typedef __attribute__((ext_vector_type(8)))  float    v8f;

#define RNN      128
#define ATT      64
#define N_ACT    10
#define IN_DIM   256
#define SIZE_TOT 65535
#define LDSS     136   // padded LDS row stride in halves (272B -> bank-shifted rows)

// f16 weight workspace layout (element offsets)
#define WS_ENC   0                         // [128][256]
#define WS_IH    (WS_ENC + 128 * 256)      // [384][128]
#define WS_HH    (WS_IH  + 384 * 128)      // [384][128]
#define WS_VW    (WS_HH  + 384 * 128)      // [64][128]
#define WS_DEC   (WS_VW  + 64 * 128)       // [16][192] (rows 10..15 zero)
#define WS_TOTAL (WS_DEC + 16 * 192)

__device__ __forceinline__ v16h combine8(v8h lo, v8h hi) {
  v16h r;
#pragma unroll
  for (int i = 0; i < 8; ++i) { r[i] = lo[i]; r[i + 8] = hi[i]; }
  return r;
}

// Fragment loader, f32 source. Row base pointer p, chunk columns c0 and c0+16
// (A/B 16-bit WMMA layout: lanes 0-15 hold K {c0..c0+7, c0+16..c0+23} with
// c0 = kb + 8*(lane>=16) handled by caller).
__device__ __forceinline__ v16h frag_f32(const float* __restrict__ p, int c0) {
  float4 a = *(const float4*)(p + c0);
  float4 b = *(const float4*)(p + c0 + 4);
  float4 c = *(const float4*)(p + c0 + 16);
  float4 d = *(const float4*)(p + c0 + 20);
  v16h r;
  r[0]  = (_Float16)a.x; r[1]  = (_Float16)a.y; r[2]  = (_Float16)a.z; r[3]  = (_Float16)a.w;
  r[4]  = (_Float16)b.x; r[5]  = (_Float16)b.y; r[6]  = (_Float16)b.z; r[7]  = (_Float16)b.w;
  r[8]  = (_Float16)c.x; r[9]  = (_Float16)c.y; r[10] = (_Float16)c.z; r[11] = (_Float16)c.w;
  r[12] = (_Float16)d.x; r[13] = (_Float16)d.y; r[14] = (_Float16)d.z; r[15] = (_Float16)d.w;
  return r;
}

// Fragment loader, f16 source (global f16 weights or LDS activation tiles).
__device__ __forceinline__ v16h frag_f16(const _Float16* __restrict__ p, int c0) {
  v8h lo = *(const v8h*)(p + c0);
  v8h hi = *(const v8h*)(p + c0 + 16);
  return combine8(lo, hi);
}

__device__ __forceinline__ float sigm(float x) {
  return 1.0f / (1.0f + __expf(-x));
}

#define WMMA(A, B, C) \
  __builtin_amdgcn_wmma_f32_16x16x32_f16(false, (A), false, (B), (short)0, (C), false, false)

__global__ void cvt_f32_to_f16_pad(const float* __restrict__ src,
                                   _Float16* __restrict__ dst,
                                   int n_src, int n_dst) {
  int i = blockIdx.x * blockDim.x + threadIdx.x;
  if (i < n_dst) dst[i] = (i < n_src) ? (_Float16)src[i] : (_Float16)0.0f;
}

__global__ __launch_bounds__(256)
void fused_critic(const float* __restrict__ obs,
                  const float* __restrict__ hidden,
                  const float* __restrict__ enc_b,
                  const float* __restrict__ b_ih,
                  const float* __restrict__ b_hh,
                  const float* __restrict__ v_b,
                  const float* __restrict__ dec_b,
                  const _Float16* __restrict__ ws,
                  float* __restrict__ out_q,
                  float* __restrict__ out_h) {
  // per-wave LDS: bufA (x tile, later v tile) + bufH (h_out tile)
  __shared__ __align__(16) _Float16 lds[8 * 2 * 16 * LDSS];

  const int tid  = threadIdx.x;
  const int warp = tid >> 5;
  const int lane = tid & 31;
  const int hs   = lane >> 4;   // K-half select (0: K..+7/+16..23, 1: +8../+24..)
  const int n16  = lane & 15;   // row index for A frags / column index for D

  _Float16* bufA = lds + warp * (2 * 16 * LDSS);
  _Float16* bufH = bufA + 16 * LDSS;

  const int row0 = (blockIdx.x * 8 + warp) * 16;
  int rowA = row0 + n16;                      // A-fragment source row (clamped)
  if (rowA > SIZE_TOT - 1) rowA = SIZE_TOT - 1;
  const int c8 = 8 * hs;

  const _Float16* w_enc = ws + WS_ENC;
  const _Float16* w_ih  = ws + WS_IH;
  const _Float16* w_hh  = ws + WS_HH;
  const _Float16* w_v   = ws + WS_VW;
  const _Float16* w_dec = ws + WS_DEC;

  // ---------------- 1) encoder: x = relu(obs @ enc_w^T + enc_b) ------------
  {
    v16h afr[8];
#pragma unroll
    for (int kt = 0; kt < 8; ++kt)
      afr[kt] = frag_f32(obs + rowA * IN_DIM, kt * 32 + c8);

    for (int nt = 0; nt < 8; ++nt) {
      v8f acc = {};
#pragma unroll
      for (int kt = 0; kt < 8; ++kt) {
        v16h b = frag_f16(w_enc + (nt * 16 + n16) * IN_DIM, kt * 32 + c8);
        acc = WMMA(afr[kt], b, acc);
      }
      const int col = nt * 16 + n16;
      const float eb = enc_b[col];
#pragma unroll
      for (int r = 0; r < 8; ++r) {
        float x = acc[r] + eb;
        x = x > 0.0f ? x : 0.0f;
        bufA[(r + 8 * hs) * LDSS + col] = (_Float16)x;   // D-layout -> LDS
      }
    }
  }
  __syncthreads();

  // ---------------- 2) GRU cell -------------------------------------------
  v16h xfr[4], hfr[4];
#pragma unroll
  for (int kt = 0; kt < 4; ++kt) {
    xfr[kt] = frag_f16(bufA + n16 * LDSS, kt * 32 + c8);          // x as A
    hfr[kt] = frag_f32(hidden + rowA * RNN, kt * 32 + c8);        // h_prev as A
  }

  for (int jt = 0; jt < 8; ++jt) {
    v8f air = {}, ahr = {}, aiz = {}, ahz = {}, ain = {}, ahn = {};
#pragma unroll
    for (int kt = 0; kt < 4; ++kt) {
      const int c0 = kt * 32 + c8;
      v16h bir = frag_f16(w_ih + (jt * 16 +   0 + n16) * RNN, c0);
      v16h biz = frag_f16(w_ih + (jt * 16 + 128 + n16) * RNN, c0);
      v16h bin = frag_f16(w_ih + (jt * 16 + 256 + n16) * RNN, c0);
      v16h bhr = frag_f16(w_hh + (jt * 16 +   0 + n16) * RNN, c0);
      v16h bhz = frag_f16(w_hh + (jt * 16 + 128 + n16) * RNN, c0);
      v16h bhn = frag_f16(w_hh + (jt * 16 + 256 + n16) * RNN, c0);
      air = WMMA(xfr[kt], bir, air);
      aiz = WMMA(xfr[kt], biz, aiz);
      ain = WMMA(xfr[kt], bin, ain);
      ahr = WMMA(hfr[kt], bhr, ahr);
      ahz = WMMA(hfr[kt], bhz, ahz);
      ahn = WMMA(hfr[kt], bhn, ahn);
    }
    const int col = jt * 16 + n16;
    const float bir_ = b_ih[col],       bhr_ = b_hh[col];
    const float biz_ = b_ih[col + 128], bhz_ = b_hh[col + 128];
    const float bin_ = b_ih[col + 256], bhn_ = b_hh[col + 256];
#pragma unroll
    for (int r = 0; r < 8; ++r) {
      const int m    = r + 8 * hs;
      const int rowc = row0 + m;
      const int rcl  = rowc < SIZE_TOT ? rowc : SIZE_TOT - 1;
      const float rg = sigm(air[r] + bir_ + ahr[r] + bhr_);
      const float zg = sigm(aiz[r] + biz_ + ahz[r] + bhz_);
      const float ng = tanhf(ain[r] + bin_ + rg * (ahn[r] + bhn_));
      const float hp = hidden[rcl * RNN + col];
      const float ho = (1.0f - zg) * ng + zg * hp;
      if (rowc < SIZE_TOT) out_h[rowc * RNN + col] = ho;   // f32 output
      bufH[m * LDSS + col] = (_Float16)ho;                 // f16 for next GEMMs
    }
  }
  __syncthreads();

  // ---------------- 3) v = relu(h_out @ v_w^T + v_b) -----------------------
  v16h hofr[4];
#pragma unroll
  for (int kt = 0; kt < 4; ++kt)
    hofr[kt] = frag_f16(bufH + n16 * LDSS, kt * 32 + c8);

  for (int nt = 0; nt < 4; ++nt) {
    v8f acc = {};
#pragma unroll
    for (int kt = 0; kt < 4; ++kt) {
      v16h b = frag_f16(w_v + (nt * 16 + n16) * RNN, kt * 32 + c8);
      acc = WMMA(hofr[kt], b, acc);
    }
    const int col = nt * 16 + n16;
    const float vb = v_b[col];
#pragma unroll
    for (int r = 0; r < 8; ++r) {
      float vv = acc[r] + vb;
      vv = vv > 0.0f ? vv : 0.0f;
      bufA[(r + 8 * hs) * LDSS + col] = (_Float16)vv;      // reuse bufA cols 0..63
    }
  }
  __syncthreads();

  // ---------------- 4) out = [h_out, v] @ dec_w^T + dec_b ------------------
  {
    v16h vfr[2];
#pragma unroll
    for (int kt = 0; kt < 2; ++kt)
      vfr[kt] = frag_f16(bufA + n16 * LDSS, kt * 32 + c8);

    v8f acc = {};
#pragma unroll
    for (int kt = 0; kt < 4; ++kt) {                       // K 0..127 : h_out
      v16h b = frag_f16(w_dec + n16 * (RNN + ATT), kt * 32 + c8);
      acc = WMMA(hofr[kt], b, acc);
    }
#pragma unroll
    for (int kt = 0; kt < 2; ++kt) {                       // K 128..191 : v
      v16h b = frag_f16(w_dec + n16 * (RNN + ATT) + RNN, kt * 32 + c8);
      acc = WMMA(vfr[kt], b, acc);
    }
    const float db = (n16 < N_ACT) ? dec_b[n16] : 0.0f;
#pragma unroll
    for (int r = 0; r < 8; ++r) {
      const int rowc = row0 + r + 8 * hs;
      if (rowc < SIZE_TOT && n16 < N_ACT)
        out_q[rowc * N_ACT + n16] = acc[r] + db;
    }
  }
}

extern "C" void kernel_launch(void* const* d_in, const int* in_sizes, int n_in,
                              void* d_out, int out_size, void* d_ws, size_t ws_size,
                              hipStream_t stream) {
  (void)in_sizes; (void)n_in; (void)out_size; (void)ws_size;

  const float* obs    = (const float*)d_in[0];
  const float* hidden = (const float*)d_in[1];
  const float* enc_w  = (const float*)d_in[2];
  const float* enc_b  = (const float*)d_in[3];
  const float* w_ih   = (const float*)d_in[4];
  const float* w_hh   = (const float*)d_in[5];
  const float* b_ih   = (const float*)d_in[6];
  const float* b_hh   = (const float*)d_in[7];
  const float* v_w    = (const float*)d_in[20];
  const float* v_b    = (const float*)d_in[21];
  const float* dec_w  = (const float*)d_in[22];
  const float* dec_b  = (const float*)d_in[23];

  _Float16* ws = (_Float16*)d_ws;

  // weight pre-conversion to f16 (dec_w zero-padded to 16 rows)
  auto cvt = [&](const float* s, _Float16* d, int nsrc, int ndst) {
    cvt_f32_to_f16_pad<<<(ndst + 255) / 256, 256, 0, stream>>>(s, d, nsrc, ndst);
  };
  cvt(enc_w, ws + WS_ENC, 128 * 256, 128 * 256);
  cvt(w_ih,  ws + WS_IH,  384 * 128, 384 * 128);
  cvt(w_hh,  ws + WS_HH,  384 * 128, 384 * 128);
  cvt(v_w,   ws + WS_VW,   64 * 128,  64 * 128);
  cvt(dec_w, ws + WS_DEC,  N_ACT * (RNN + ATT), 16 * (RNN + ATT));

  float* out_q = (float*)d_out;                    // [65535,10]
  float* out_h = out_q + SIZE_TOT * N_ACT;         // [65535,128]

  const int rows_per_block = 8 * 16;               // 8 waves * 16 rows
  const int grid = (SIZE_TOT + rows_per_block - 1) / rows_per_block;  // 512
  fused_critic<<<grid, 256, 0, stream>>>(obs, hidden, enc_b, b_ih, b_hh,
                                         v_b, dec_b, ws, out_q, out_h);
}
